// LayerNormLSTMCell_18511309046399
// MI455X (gfx1250) — matrix operand order repaired and checked by reading, not compiled, not run on hardware
//
#include <hip/hip_runtime.h>
#include <hip/hip_bf16.h>
#include <math.h>

// ---------------------------------------------------------------------------
// LayerNorm LSTM cell for MI455X (gfx1250), wave32 + WMMA bf16 + async LDS.
//   B=16384, IN=H=1024, 4H=4096
// One workgroup owns a 16-row stripe. Activation tile (16x1024 bf16) is
// staged to LDS with GLOBAL_LOAD_ASYNC_TO_LDS_B128 (ASYNCcnt). WMMA uses
// weights as A (global b128 pairs) and activations as B (ds b128 pairs),
// 4 output tiles per wave iteration -> 2.5 b128 loads per v_wmma.
// Both full 16x4096 logit tiles live in LDS (CDNA5 320KB/WGP).
// ---------------------------------------------------------------------------

typedef __attribute__((ext_vector_type(16))) __bf16 v16bf;
typedef __attribute__((ext_vector_type(8)))  float  v8f;

#define B_ROWS  16384
#define K_DIM   1024
#define H_DIM   1024
#define O_DIM   4096
#define M_TILE  16
#define A_PITCH 1032       // halves/row in LDS stage: 2064B = 516 dw, 516%64=4
#define LN_EPS  1e-5f

// ---- fp32 -> bf16 conversion pre-pass (vectorized) ------------------------
__global__ __launch_bounds__(256) void cvt_f32_to_bf16_k(
    const float4* __restrict__ in, uint2* __restrict__ out, int n4) {
  int i = blockIdx.x * blockDim.x + threadIdx.x;
  int stride = gridDim.x * blockDim.x;
  for (; i < n4; i += stride) {
    float4 v = in[i];
    union { __bf16 h[4]; uint2 u; } p;
    p.h[0] = (__bf16)v.x; p.h[1] = (__bf16)v.y;
    p.h[2] = (__bf16)v.z; p.h[3] = (__bf16)v.w;
    out[i] = p.u;
  }
}

__device__ __forceinline__ float sigf(float x) {
  return 1.0f / (1.0f + __expf(-x));
}

// ---- fused GEMM + LayerNorm + LSTM pointwise ------------------------------
__global__ __launch_bounds__(256) void lnlstm_fused_k(
    const __bf16* __restrict__ xb,  const __bf16* __restrict__ hb,
    const __bf16* __restrict__ wib, const __bf16* __restrict__ whb,
    const float*  __restrict__ c_old,
    const float*  __restrict__ gi, const float* __restrict__ bi,
    const float*  __restrict__ gh, const float* __restrict__ bh,
    const float*  __restrict__ gc, const float* __restrict__ bc,
    const float*  __restrict__ bias,
    float* __restrict__ h_new, float* __restrict__ c_new) {

  // ~295 KB static LDS (CDNA5: 320 KB per workgroup)
  __shared__ __bf16 Li[M_TILE * O_DIM];        // raw i2h logits, 128 KB
  __shared__ __bf16 Lh[M_TILE * O_DIM];        // raw h2h logits, 128 KB
  __shared__ __bf16 As[M_TILE * A_PITCH];      // staged activation tile, 32.25 KB

  const int tid  = threadIdx.x;
  const int wave = tid >> 5;
  const int lane = tid & 31;
  const int lhf  = lane >> 4;                  // half-wave select
  const int ll   = lane & 15;
  const int rowBase = blockIdx.x * M_TILE;

  const unsigned asBase = (unsigned)(uintptr_t)(&As[0]); // LDS byte offset

  // -- stage 16x1024 bf16 activation tile into LDS via async DMA loads ----
  auto stage_act = [&](const __bf16* __restrict__ act) {
#pragma unroll
    for (int it = 0; it < 8; ++it) {
      int lin = tid + it * 256;                // 0..2047 16B-chunks
      int rr  = lin >> 7;                      // row 0..15
      int cc  = (lin & 127) << 3;              // half-offset 0..1016
      unsigned lds = asBase + (unsigned)((rr * A_PITCH + cc) * 2);
      unsigned long long ga =
          (unsigned long long)(act + (size_t)(rowBase + rr) * K_DIM + cc);
      asm volatile("global_load_async_to_lds_b128 %0, %1, off"
                   :: "v"(lds), "v"(ga) : "memory");
    }
    asm volatile("s_wait_asynccnt 0x0" ::: "memory");
  };

  // -- WMMA GEMM: Ldst[16 x 4096] = As[16xK] * wgt^T ----------------------
  // A-frag = weight tile 16x32 (global b128 pairs), B-frag = activation
  // 32x16 (ds b128 pairs, k = 8*lhf+[0,8) and 16+8*lhf+[0,8)). 4 tiles/iter.
  auto gemm_to_lds = [&](const __bf16* __restrict__ wgt,
                         __bf16* __restrict__ Ldst) {
    const __bf16* bbase = &As[ll * A_PITCH + lhf * 8];
    for (int g = 0; g < 8; ++g) {              // wave-uniform
      const int ntb = g * 32 + wave * 4;       // 4 consecutive col-tiles
      const __bf16* w0 = wgt + (size_t)(ntb * 16 + ll) * K_DIM + lhf * 8;
      const __bf16* w1 = w0 + (size_t)16 * K_DIM;
      const __bf16* w2 = w1 + (size_t)16 * K_DIM;
      const __bf16* w3 = w2 + (size_t)16 * K_DIM;
      v8f acc0 = {0.f,0.f,0.f,0.f,0.f,0.f,0.f,0.f};
      v8f acc1 = acc0, acc2 = acc0, acc3 = acc0;
#pragma unroll 2
      for (int k = 0; k < K_DIM; k += 32) {
        union { v16bf f; uint4 u[2]; } b, a0, a1, a2, a3;
        b.u[0]  = *(const uint4*)(bbase + k);
        b.u[1]  = *(const uint4*)(bbase + k + 16);
        a0.u[0] = *(const uint4*)(w0 + k); a0.u[1] = *(const uint4*)(w0 + k + 16);
        a1.u[0] = *(const uint4*)(w1 + k); a1.u[1] = *(const uint4*)(w1 + k + 16);
        a2.u[0] = *(const uint4*)(w2 + k); a2.u[1] = *(const uint4*)(w2 + k + 16);
        a3.u[0] = *(const uint4*)(w3 + k); a3.u[1] = *(const uint4*)(w3 + k + 16);
        acc0 = __builtin_amdgcn_wmma_f32_16x16x32_bf16(false, a0.f, false, b.f, (short)0, acc0, false, false);
        acc1 = __builtin_amdgcn_wmma_f32_16x16x32_bf16(false, a1.f, false, b.f, (short)0, acc1, false, false);
        acc2 = __builtin_amdgcn_wmma_f32_16x16x32_bf16(false, a2.f, false, b.f, (short)0, acc2, false, false);
        acc3 = __builtin_amdgcn_wmma_f32_16x16x32_bf16(false, a3.f, false, b.f, (short)0, acc3, false, false);
      }
      // C layout: M = out-col = v + 8*lhf, N = batch row = ll ->
      // lane writes 8 contiguous bf16 = one ds_store_b128.
      auto store_tile = [&](const v8f& acc, int tile) {
        union { __bf16 h[8]; uint4 u; } p;
#pragma unroll
        for (int v = 0; v < 8; ++v) p.h[v] = (__bf16)acc[v];
        *(uint4*)&Ldst[(size_t)ll * O_DIM + tile * 16 + lhf * 8] = p.u;
      };
      store_tile(acc0, ntb + 0);
      store_tile(acc1, ntb + 1);
      store_tile(acc2, ntb + 2);
      store_tile(acc3, ntb + 3);
    }
  };

  stage_act(xb);
  __syncthreads();
  gemm_to_lds(wib, Li);
  __syncthreads();                 // all waves done reading As
  stage_act(hb);
  __syncthreads();
  gemm_to_lds(whb, Lh);
  __syncthreads();                 // Li/Lh complete across waves

  // ---------------- per-row LayerNorm statistics (register-resident) -----
  const int r  = tid >> 4;         // 16 threads per row, contiguous lanes
  const int c0 = tid & 15;

  auto row_stats = [&](const __bf16* __restrict__ L, float& mu, float& rs) {
    float s = 0.f, q = 0.f;
    for (int j = c0; j < O_DIM; j += 16) {
      float v = (float)L[r * O_DIM + j];
      s += v; q += v * v;
    }
#pragma unroll
    for (int m = 8; m; m >>= 1) {  // butterfly: every lane gets the total
      s += __shfl_xor(s, m, 32);
      q += __shfl_xor(q, m, 32);
    }
    mu = s * (1.f / O_DIM);
    rs = rsqrtf(q * (1.f / O_DIM) - mu * mu + LN_EPS);
  };

  float muI, rsI, muH, rsH;
  row_stats(Li, muI, rsI);
  row_stats(Lh, muH, rsH);

  const size_t grow = (size_t)(rowBase + r) * H_DIM;

  auto logit = [&](int col) {
    float vi = ((float)Li[r * O_DIM + col] - muI) * rsI * gi[col] + bi[col];
    float vh = ((float)Lh[r * O_DIM + col] - muH) * rsH * gh[col] + bh[col];
    return vi + vh + bias[col];
  };

  // ---------------- gates: c_new = sig(f)*c + sig(i)*tanh(g) -------------
  float s = 0.f, q = 0.f;
  for (int j = c0; j < H_DIM; j += 16) {
    float lf = logit(j);                 // f: cols [0, H)
    float li = logit(H_DIM + j);         // i: cols [H, 2H)
    float lg = logit(3 * H_DIM + j);     // g: cols [3H, 4H)
    float cn = sigf(lf) * c_old[grow + j] + sigf(li) * tanhf(lg);
    c_new[grow + j] = cn;
    s += cn; q += cn * cn;
  }
#pragma unroll
  for (int m = 8; m; m >>= 1) {
    s += __shfl_xor(s, m, 32);
    q += __shfl_xor(q, m, 32);
  }
  const float muC = s * (1.f / H_DIM);
  const float rsC = rsqrtf(q * (1.f / H_DIM) - muC * muC + LN_EPS);

  // ---------------- h_new = sig(o) * tanh(LN_c(c_new)) -------------------
  for (int j = c0; j < H_DIM; j += 16) {
    float lo = logit(2 * H_DIM + j);     // o: cols [2H, 3H)
    float cn = c_new[grow + j];          // same-thread store->load, in order
    float ln = (cn - muC) * rsC * gc[j] + bc[j];
    h_new[grow + j] = sigf(lo) * tanhf(ln);
  }
}

// ---------------------------------------------------------------------------
extern "C" void kernel_launch(void* const* d_in, const int* in_sizes, int n_in,
                              void* d_out, int out_size, void* d_ws, size_t ws_size,
                              hipStream_t stream) {
  const float* x    = (const float*)d_in[0];
  const float* h    = (const float*)d_in[1];
  const float* c    = (const float*)d_in[2];
  const float* Wi   = (const float*)d_in[3];
  const float* Wh   = (const float*)d_in[4];
  const float* gi   = (const float*)d_in[5];
  const float* bi   = (const float*)d_in[6];
  const float* gh   = (const float*)d_in[7];
  const float* bh   = (const float*)d_in[8];
  const float* gc   = (const float*)d_in[9];
  const float* bc   = (const float*)d_in[10];
  const float* bias = (const float*)d_in[11];

  float* out   = (float*)d_out;
  float* h_new = out;                                   // first output
  float* c_new = out + (size_t)B_ROWS * H_DIM;          // second output

  // ws layout (bf16 copies): xb 32MB | hb 32MB | wib 8MB | whb 8MB = 80MB
  char* ws = (char*)d_ws;
  __bf16* xb  = (__bf16*)(ws);
  __bf16* hb  = (__bf16*)(ws + (size_t)33554432);
  __bf16* wib = (__bf16*)(ws + (size_t)67108864);
  __bf16* whb = (__bf16*)(ws + (size_t)75497472);

  const int nAct4 = B_ROWS * K_DIM / 4;   // 4.2M float4
  const int nW4   = O_DIM * K_DIM / 4;    // 1.05M float4
  cvt_f32_to_bf16_k<<<4096, 256, 0, stream>>>((const float4*)x,  (uint2*)xb,  nAct4);
  cvt_f32_to_bf16_k<<<4096, 256, 0, stream>>>((const float4*)h,  (uint2*)hb,  nAct4);
  cvt_f32_to_bf16_k<<<2048, 256, 0, stream>>>((const float4*)Wi, (uint2*)wib, nW4);
  cvt_f32_to_bf16_k<<<2048, 256, 0, stream>>>((const float4*)Wh, (uint2*)whb, nW4);

  lnlstm_fused_k<<<B_ROWS / M_TILE, 256, 0, stream>>>(
      xb, hb, wib, whb, c, gi, bi, gh, bh, gc, bc, bias, h_new, c_new);
}
